// Encoder_47528108097609
// MI455X (gfx1250) — matrix operand order repaired and checked by reading, not compiled, not run on hardware
//
#include <hip/hip_runtime.h>

// ---------------------------------------------------------------------------
// MI455X / gfx1250 implementation.
//   - dist encoder layer-2 GEMM (40 GFLOP) on v_wmma_f32_16x16x32_f16
//   - point-encoder MLP layers 2..4 on WMMA as well (LDS-staged fragments)
//   - all masked outputs fused into the GEMM epilogues (629 MB stores total)
//   - dist scalars double-buffered via GLOBAL_LOAD_ASYNC_TO_LDS_B32 +
//     s_wait_asynccnt (ASYNCcnt) so the H-fill never waits on global memory
// ---------------------------------------------------------------------------

typedef __attribute__((ext_vector_type(16))) _Float16 v16h;
typedef __attribute__((ext_vector_type(8)))  _Float16 v8h;
typedef __attribute__((ext_vector_type(8)))  float    v8f;

#define Bc    4
#define Ac    3
#define Dc    3
#define SZc   160
#define HWc   25600          // SZ*SZ
#define FDc   256
#define NCOL  153603         // HW + D*HW + A + HW + HW
#define COL_DIST 25600
#define COL_AG   102400
#define COL_PH   102403
#define COL_GH   128003
#define PTOT  307200         // B*D*HW

#if defined(__AMDGCN__) && __has_builtin(__builtin_amdgcn_global_load_async_to_lds_b32)
#define HAVE_ASYNC_LDS 1
#else
#define HAVE_ASYNC_LDS 0
#endif

typedef __attribute__((address_space(1))) int* gptr_i32;  // clang: "__device__ int *"
typedef __attribute__((address_space(3))) int* lptr_i32;

static __device__ __forceinline__ void stage_x_async(const float* g, float* l) {
#if HAVE_ASYNC_LDS
  __builtin_amdgcn_global_load_async_to_lds_b32((gptr_i32)(void*)(float*)g,
                                                (lptr_i32)(void*)l, 0, 0);
#else
  *l = *g;  // synchronous fallback
#endif
}
static __device__ __forceinline__ void wait_async_all() {
#if HAVE_ASYNC_LDS
  asm volatile("s_wait_asynccnt 0x0" ::: "memory");
#endif
}

static __device__ __forceinline__ v8f wmma_f16(v16h a, v16h b, v8f c) {
  // D = A(16x32 f16) * B(32x16 f16) + C(16x16 f32)
  return __builtin_amdgcn_wmma_f32_16x16x32_f16(false, a, false, b,
                                                (short)0, c, false, false);
}

// A fragment: lane holds row M = lane%16.  Per ISA table (16-bit A 16x32):
//   half = lane/16;  element e<8  -> K = kbase + 8*half + e
//                    element e>=8 -> K = kbase + 8*half + 16 + (e-8)
static __device__ __forceinline__ v16h load_A_g(const float* __restrict__ W,
                                                int row, int in_dim,
                                                int kbase, int half) {
  const float* wr = W + row * in_dim + kbase + 8 * half;
  v16h a;
#pragma unroll
  for (int e = 0; e < 8; ++e) a[e] = (_Float16)wr[e];
#pragma unroll
  for (int e = 0; e < 8; ++e) a[8 + e] = (_Float16)wr[16 + e];
  return a;
}

// B fragment from LDS tile stored as Hc[point][k] (k contiguous):
//   lane: n = lane%16, half = lane/16; koff = ks*32 + 16*half; element e<8 ->
//   K = ks*32 + 8*half + e, e>=8 -> K = ks*32 + 16 + 8*half + (e-8).
static __device__ __forceinline__ v16h load_B_lds(const _Float16* __restrict__ buf,
                                                  int row_e, int n, int koff) {
  const _Float16* p = buf + n * row_e + koff - (koff & 31) / 2;
  v8h lo = *(const v8h*)(p);
  v8h hi = *(const v8h*)(p + 16);
  v16h r;
#pragma unroll
  for (int e = 0; e < 8; ++e) { r[e] = lo[e]; r[8 + e] = hi[e]; }
  return r;
}

// One hidden MLP layer (relu) as WMMA GEMM: out[m,n] = relu(W[m,:]@in[:,n]+b[m])
// C layout: lane n = lane%16, VGPR v -> m = mb*16 + 8*half + v (contiguous in v
// -> one 16B v8h store per m-block).
template <int IN, int OUT, int ROW_E>
static __device__ __forceinline__ void layer_mid(const float* __restrict__ W,
                                                 const float* __restrict__ bias,
                                                 const _Float16* __restrict__ inb,
                                                 _Float16* __restrict__ outb,
                                                 int lane) {
  const int n = lane & 15, half = lane >> 4;
#pragma unroll 1
  for (int mb = 0; mb < OUT / 16; ++mb) {   // keep pressure low: 1 A-frag live
    v8f acc;
#pragma unroll
    for (int v = 0; v < 8; ++v) acc[v] = bias[mb * 16 + 8 * half + v];
#pragma unroll
    for (int ks = 0; ks < IN / 32; ++ks) {
      v16h a = load_A_g(W, mb * 16 + n, IN, ks * 32, half);
      v16h b = load_B_lds(inb, ROW_E, n, ks * 32 + 16 * half);
      acc = wmma_f16(a, b, acc);
    }
    v8h o;
#pragma unroll
    for (int v = 0; v < 8; ++v) o[v] = (_Float16)fmaxf(acc[v], 0.0f);
    *(v8h*)(outb + n * ROW_E + mb * 16 + 8 * half) = o;
  }
}

// ---------------------------------------------------------------------------
// Kernel 1: grid-point encoder. One wave per 16-point tile; 2 passes
// (frontier-type, history-type). Final 256-wide layer fuses masked scatter of
// frontier_out / gh_out (pass 0) and ph_out (pass 1) for all 4 batches.
// ---------------------------------------------------------------------------
#define ROWE_PT 136  // 128 f16 payload + pad, 272B (16B aligned, bank-spread)

__global__ void point_mlp_kernel(const int* __restrict__ inputs,
                                 const int* __restrict__ pos_hist,
                                 const int* __restrict__ goal_hist,
                                 const float* __restrict__ W0, const float* __restrict__ b0,
                                 const float* __restrict__ W1, const float* __restrict__ b1,
                                 const float* __restrict__ W2, const float* __restrict__ b2,
                                 const float* __restrict__ W3, const float* __restrict__ b3,
                                 float* __restrict__ out) {
  extern __shared__ _Float16 smem[];
  const int lane = threadIdx.x & 31;
  const int wave = threadIdx.x >> 5;            // 4 waves / WG
  const int n = lane & 15, half = lane >> 4;
  _Float16* buf0 = smem + wave * 2 * 16 * ROWE_PT;
  _Float16* buf1 = buf0 + 16 * ROWE_PT;
  const int tile = blockIdx.x * 4 + wave;       // 400 blocks * 4 = 1600 tiles
  const int p = tile * 16 + n;                  // this lane's grid point
  const float nx = (float)((p / SZc) - (SZc / 2)) * (1.0f / (SZc * 0.7f));
  const float ny = (float)((p % SZc) - (SZc / 2)) * (1.0f / (SZc * 0.7f));

#pragma unroll 1
  for (int et = 0; et < 2; ++et) {              // 0: frontier pts, 1: history pts
    const float c2 = (et == 0) ? 1.0f : 0.0f;
    const float c3 = 1.0f - c2;
    // Layer 1 (4 -> 32) on VALU: lanes split the 32 outputs per point.
#pragma unroll
    for (int oo = 0; oo < 16; ++oo) {
      int o = half * 16 + oo;
      float v = W0[o * 4 + 0] * nx + W0[o * 4 + 1] * ny +
                W0[o * 4 + 2] * c2 + W0[o * 4 + 3] * c3 + b0[o];
      buf0[n * ROWE_PT + o] = (_Float16)fmaxf(v, 0.0f);
    }
    __syncthreads();
    layer_mid<32, 64, ROWE_PT>(W1, b1, buf0, buf1, lane);    // 32 -> 64
    __syncthreads();
    layer_mid<64, 128, ROWE_PT>(W2, b2, buf1, buf0, lane);   // 64 -> 128
    __syncthreads();

    // Final layer (128 -> 256), no relu; fused masked scatter.
    float m0[Bc], m1[Bc];
#pragma unroll
    for (int b = 0; b < Bc; ++b) {
      if (et == 0) {
        m0[b] = (inputs[(b * 2 + 1) * HWc + p] > 0) ? 1.0f : 0.0f;  // fmask
        m1[b] = (goal_hist[b * HWc + p] > 0) ? 1.0f : 0.0f;         // gmask
      } else {
        m0[b] = (pos_hist[b * HWc + p] > 0) ? 1.0f : 0.0f;          // pmask
        m1[b] = 0.0f;
      }
    }
#pragma unroll 1
    for (int mb = 0; mb < 16; ++mb) {           // unroll(1): avoid A-frag spills
      v8f acc;
#pragma unroll
      for (int v = 0; v < 8; ++v) acc[v] = b3[mb * 16 + 8 * half + v];
#pragma unroll
      for (int ks = 0; ks < 4; ++ks) {
        v16h a = load_A_g(W3, mb * 16 + n, 128, ks * 32, half);
        v16h b = load_B_lds(buf0, ROWE_PT, n, ks * 32 + 16 * half);
        acc = wmma_f16(a, b, acc);
      }
#pragma unroll
      for (int v = 0; v < 8; ++v) {
        int i = mb * 16 + 8 * half + v;
        float val = acc[v];
        if (et == 0) {
#pragma unroll
          for (int b = 0; b < Bc; ++b) {
            float* r = out + (size_t)(b * FDc + i) * NCOL;
            r[p] = val * m0[b];             // frontier_out
            r[COL_GH + p] = val * m1[b];    // gh_out (frontier feats, gmask)
          }
        } else {
#pragma unroll
          for (int b = 0; b < Bc; ++b)
            out[(size_t)(b * FDc + i) * NCOL + COL_PH + p] = val * m0[b];
        }
      }
    }
    __syncthreads();  // buf0 reused next pass
  }
}

// ---------------------------------------------------------------------------
// Kernel 2: dist encoder. 512 threads = 16 waves; wave w owns output rows
// [16w, 16w+16). W2 A-fragments (8 K-steps) live in VGPRs for the whole
// kernel. Per 16-point tile the WG builds H (256 x 16 f16) in LDS from the
// scalar layer-1, then each wave runs 8 WMMAs and stores masked results.
// The 16 dist scalars per tile are double-buffered through LDS with
// GLOBAL_LOAD_ASYNC_TO_LDS_B32, drained by s_wait_asynccnt before the
// publishing barrier.
// ---------------------------------------------------------------------------
#define ROWE_D 264  // 256 f16 payload + pad = 528B rows

__global__ void dist_kernel(const float* __restrict__ dist,
                            const int* __restrict__ inputs,
                            const float* __restrict__ Wd0, const float* __restrict__ bd0,
                            const float* __restrict__ Wd1, const float* __restrict__ bd1,
                            float* __restrict__ out) {
  __shared__ _Float16 Hc[16 * ROWE_D];
  __shared__ float xstage[2][16];      // async-staged dist scalars (ping-pong)
  const int lane = threadIdx.x & 31;
  const int wave = threadIdx.x >> 5;   // 0..15 -> M-block
  const int n = lane & 15, half = lane >> 4;
  const int t = threadIdx.x;

  // Loop-invariant A fragments: W2 rows 16*wave .. 16*wave+15 (f32 -> f16).
  v16h A[8];
#pragma unroll
  for (int ks = 0; ks < 8; ++ks)
    A[ks] = load_A_g(Wd1, wave * 16 + n, 256, ks * 32, half);
  v8f cbias;
#pragma unroll
  for (int v = 0; v < 8; ++v) cbias[v] = bd1[wave * 16 + 8 * half + v];

  const int ntiles = PTOT / 16;                       // 19200
  // Prologue: stage this block's first tile.
  if (t < 16) stage_x_async(dist + blockIdx.x * 16 + t, &xstage[0][t]);

  int cur = 0;
#pragma unroll 1
  for (int tile = blockIdx.x; tile < ntiles; tile += gridDim.x) {
    wait_async_all();   // wave0: xstage[cur] landed in LDS (others: ASYNCcnt==0)
    __syncthreads();    // publish xstage[cur]; Hc free for refill
    const int nxt = tile + gridDim.x;
    if (nxt < ntiles && t < 16)
      stage_x_async(dist + nxt * 16 + t, &xstage[cur ^ 1][t]);  // overlap w/ WMMAs
    {
      // H fill: thread t covers point (t&15), j in [8*(t>>4), 8*(t>>4)+8)
      const int pn = t & 15, jb = (t >> 4) * 8;
      const float x = xstage[cur][pn];
      v8h hv;
#pragma unroll
      for (int e = 0; e < 8; ++e) {
        int j = jb + e;
        hv[e] = (_Float16)fmaxf(Wd0[j] * x + bd0[j], 0.0f);
      }
      *(v8h*)(&Hc[pn * ROWE_D + jb]) = hv;
    }
    __syncthreads();

    v8f acc = cbias;
#pragma unroll
    for (int ks = 0; ks < 8; ++ks)
      acc = wmma_f16(A[ks], load_B_lds(Hc, ROWE_D, n, ks * 32 + 16 * half), acc);

    // Mask (fmask broadcast over D) and store into dist slice.
    const int p = tile * 16 + n;
    const int b = p / (Dc * HWc);
    const int within = p - b * (Dc * HWc);
    const int hw = within % HWc;
    const float m = (inputs[(b * 2 + 1) * HWc + hw] > 0) ? 1.0f : 0.0f;
    if (nxt < ntiles) {  // prefetch next tile's mask word (global_prefetch_b8)
      const int p2 = nxt * 16 + n;
      const int b2 = p2 / (Dc * HWc);
      const int w2 = p2 - b2 * (Dc * HWc);
      __builtin_prefetch(inputs + (b2 * 2 + 1) * HWc + (w2 % HWc), 0, 0);
    }
    float* ob = out + (size_t)(b * FDc) * NCOL + COL_DIST + within;
#pragma unroll
    for (int v = 0; v < 8; ++v) {
      int i = wave * 16 + 8 * half + v;
      ob[(size_t)i * NCOL] = acc[v] * m;
    }
    cur ^= 1;
  }
}

// ---------------------------------------------------------------------------
// Kernel 3: agent points (B*A = 12): tiny LDS-staged VALU MLP, single WG.
// ---------------------------------------------------------------------------
__global__ void agent_kernel(const int* __restrict__ extras,
                             const float* __restrict__ W0, const float* __restrict__ b0,
                             const float* __restrict__ W1, const float* __restrict__ b1,
                             const float* __restrict__ W2, const float* __restrict__ b2,
                             const float* __restrict__ W3, const float* __restrict__ b3,
                             float* __restrict__ out) {
  __shared__ float abuf0[256];
  __shared__ float abuf1[256];
  const int t = threadIdx.x;  // 256 threads
#pragma unroll 1
  for (int pt = 0; pt < Bc * Ac; ++pt) {
    const int b = pt / Ac, a = pt % Ac;
    const float x = (float)(extras[(b * Ac + a) * 2 + 0] - (SZc / 2)) * (1.0f / (SZc * 0.7f));
    const float y = (float)(extras[(b * Ac + a) * 2 + 1] - (SZc / 2)) * (1.0f / (SZc * 0.7f));
    if (t < 32)   // ch2 = 0, ch3 = 1 (history-type point)
      abuf0[t] = fmaxf(W0[t * 4 + 0] * x + W0[t * 4 + 1] * y + W0[t * 4 + 3] + b0[t], 0.0f);
    __syncthreads();
    if (t < 64) {
      float s = b1[t];
      for (int i = 0; i < 32; ++i) s += W1[t * 32 + i] * abuf0[i];
      abuf1[t] = fmaxf(s, 0.0f);
    }
    __syncthreads();
    if (t < 128) {
      float s = b2[t];
      for (int i = 0; i < 64; ++i) s += W2[t * 64 + i] * abuf1[i];
      abuf0[t] = fmaxf(s, 0.0f);
    }
    __syncthreads();
    {
      float s = b3[t];
      for (int i = 0; i < 128; ++i) s += W3[t * 128 + i] * abuf0[i];
      out[(size_t)(b * FDc + t) * NCOL + COL_AG + a] = s;
    }
    __syncthreads();
  }
}

// ---------------------------------------------------------------------------
extern "C" void kernel_launch(void* const* d_in, const int* in_sizes, int n_in,
                              void* d_out, int out_size, void* d_ws, size_t ws_size,
                              hipStream_t stream) {
  (void)in_sizes; (void)n_in; (void)out_size; (void)d_ws; (void)ws_size;
  // setup_inputs() dict order, tuples flattened in order:
  const int*   inputs    = (const int*)  d_in[0];
  const float* dist      = (const float*)d_in[1];
  const int*   pos_hist  = (const int*)  d_in[2];
  const int*   goal_hist = (const int*)  d_in[3];
  const int*   extras    = (const int*)  d_in[4];
  const float* W0 = (const float*)d_in[5];   // (32,4)
  const float* W1 = (const float*)d_in[6];   // (64,32)
  const float* W2 = (const float*)d_in[7];   // (128,64)
  const float* W3 = (const float*)d_in[8];   // (256,128)
  const float* b0 = (const float*)d_in[9];
  const float* b1 = (const float*)d_in[10];
  const float* b2 = (const float*)d_in[11];
  const float* b3 = (const float*)d_in[12];
  const float* Wd0 = (const float*)d_in[13]; // (256,1)
  const float* Wd1 = (const float*)d_in[14]; // (256,256)
  const float* bd0 = (const float*)d_in[15];
  const float* bd1 = (const float*)d_in[16];
  float* out = (float*)d_out;

  const size_t smem_pt = (size_t)4 /*waves*/ * 2 * 16 * ROWE_PT * sizeof(_Float16);
  point_mlp_kernel<<<400, 128, smem_pt, stream>>>(inputs, pos_hist, goal_hist,
                                                  W0, b0, W1, b1, W2, b2, W3, b3, out);
  dist_kernel<<<2400, 512, 0, stream>>>(dist, inputs, Wd0, bd0, Wd1, bd1, out);
  agent_kernel<<<1, 256, 0, stream>>>(extras, W0, b0, W1, b1, W2, b2, W3, b3, out);
}
